// Decoder_AUTOTRANS_89824946029072
// MI455X (gfx1250) — matrix operand order repaired and checked by reading, not compiled, not run on hardware
//
#include <hip/hip_runtime.h>
#include <math.h>

// ---------------- constants ----------------
#define D_   512
#define H_   8
#define L_   8
#define F_   2048
#define NJ_  25
#define NF_  6
#define NC_  12
#define B_   64
#define T_   256
#define NT_  (B_*T_)      // 16384 rows
#define DK_  64
#define EPS_ 1e-6f

typedef _Float16 v16h __attribute__((ext_vector_type(16)));
typedef float    v8f  __attribute__((ext_vector_type(8)));

__device__ __forceinline__ v8f wmma16(const v16h a, const v16h b, const v8f c) {
  // D = A(16x32 f16) * B(32x16 f16) + C(16x16 f32)
  return __builtin_amdgcn_wmma_f32_16x16x32_f16(false, a, false, b, (short)0, c, false, false);
}

__device__ __forceinline__ unsigned lds_addr_of(const void* p) {
  // generic -> LDS(addrspace 3) -> 32-bit wave-relative LDS byte offset
  return (unsigned)(unsigned long long)(__attribute__((address_space(3))) const void*)p;
}

// =====================================================================
// GEMM: C[M,N] = A[M,K](f32) @ BT[N,K](f16, pre-transposed) (+bias)(+res)(relu)
// 128x128 block tile, 8 waves, each wave 32x64 -> 8 WMMAs per K-chunk.
// B tile staged into LDS with global_load_async_to_lds_b128 (ASYNCcnt).
// Requires K % 32 == 0, M % 128 == 0. N-edge guarded.
// =====================================================================
__launch_bounds__(256)
__global__ void gemm_wmma(const float* __restrict__ A, const _Float16* __restrict__ BT,
                          const float* __restrict__ bias, const float* __restrict__ res,
                          float* __restrict__ C, int M, int N, int K, int relu)
{
  __shared__ _Float16 sA[128][34];      // [m][k], pad->conflict-free pair reads
  __shared__ _Float16 sB[128 * 40];     // [n][k], 80B row stride (16B aligned)
  const int tid    = threadIdx.x;
  const int lane   = tid & 31;
  const int w      = tid >> 5;
  const int lanelo = lane & 15, lanehi = lane >> 4;
  const int wr = w >> 1, wc = w & 1;    // 4x2 wave grid
  const int tM = blockIdx.y * 128;
  const int tN = blockIdx.x * 128;

  v8f c[2][4];
  #pragma unroll
  for (int i = 0; i < 2; ++i)
    #pragma unroll
    for (int s = 0; s < 4; ++s) c[i][s] = (v8f){0,0,0,0,0,0,0,0};

  const int ar = tid >> 1;              // A row 0..127
  const int ac = (tid & 1) * 16;        // 16 consecutive k
  const unsigned sBbase = lds_addr_of(&sB[0]);

  for (int k0 = 0; k0 < K; k0 += 32) {
    __syncthreads();
    // ---- B tile: async f16 copy, 2 x 16B segments per thread ----
    #pragma unroll
    for (int j = 0; j < 2; ++j) {
      const int seg  = tid * 2 + j;     // 0..511
      const int row  = seg >> 2;        // n within tile
      const int part = seg & 3;         // 16B segment within 64B row
      int ng = tN + row; if (ng > N - 1) ng = N - 1;   // clamp (edge tiles)
      const unsigned lo = sBbase + (unsigned)(row * 80 + part * 16);
      const unsigned long long ga =
          (unsigned long long)(const void*)(BT + (size_t)ng * K + k0 + part * 8);
      asm volatile("global_load_async_to_lds_b128 %0, %1, off"
                   :: "v"(lo), "v"(ga) : "memory");
    }
    // ---- A tile: f32 -> f16 through VGPRs ----
    {
      const float* ap = A + (size_t)(tM + ar) * K + k0 + ac;
      #pragma unroll
      for (int i = 0; i < 16; ++i) sA[ar][ac + i] = (_Float16)ap[i];
      if (k0 + 32 < K) __builtin_prefetch(ap + 32, 0, 1);
    }
    asm volatile("s_wait_asynccnt 0x0" ::: "memory");
    __syncthreads();

    // ---- fragments + 8 WMMAs ----
    v16h a0, a1;
    #pragma unroll
    for (int v = 0; v < 8; ++v) {
      const int kb = (lanehi << 4) + (v << 1);
      a0[2*v] = sA[wr*32 + lanelo][kb];       a0[2*v+1] = sA[wr*32 + lanelo][kb+1];
      a1[2*v] = sA[wr*32 + 16 + lanelo][kb];  a1[2*v+1] = sA[wr*32 + 16 + lanelo][kb+1];
    }
    #pragma unroll
    for (int s = 0; s < 4; ++s) {
      v16h bf;
      const int n = wc * 64 + s * 16 + lanelo;
      #pragma unroll
      for (int v = 0; v < 8; ++v) {
        const int kb = (lanehi << 4) + (v << 1);
        bf[2*v] = sB[n * 40 + kb];  bf[2*v+1] = sB[n * 40 + kb + 1];
      }
      c[0][s] = wmma16(a0, bf, c[0][s]);
      c[1][s] = wmma16(a1, bf, c[1][s]);
    }
  }

  // ---- epilogue ----
  #pragma unroll
  for (int i = 0; i < 2; ++i)
    #pragma unroll
    for (int s = 0; s < 4; ++s)
      #pragma unroll
      for (int v = 0; v < 8; ++v) {
        const int row = tM + wr*32 + i*16 + v + (lanehi << 3);
        const int col = tN + wc*64 + s*16 + lanelo;
        if (col < N && row < M) {
          float x = c[i][s][v];
          if (bias) x += bias[col];
          if (res)  x += res[(size_t)row * N + col];
          if (relu) x = fmaxf(x, 0.f);
          C[(size_t)row * N + col] = x;
        }
      }
}

// =====================================================================
// Weight convert + transpose: WT[g][n][k] = (f16) W[g][k][n]
// =====================================================================
__global__ void wtrans_k(const float* __restrict__ W, _Float16* __restrict__ WT,
                         int K, int N)
{
  const size_t KN = (size_t)K * N;
  const size_t g  = blockIdx.y;
  const size_t o  = (size_t)blockIdx.x * 256 + threadIdx.x;   // < KN
  const int n = (int)(o / K);
  const int k = (int)(o % K);
  WT[g * KN + o] = (_Float16)W[g * KN + (size_t)k * N + n];
}

// =====================================================================
// Fused flash-attention (WMMA). Q,K,V: [B*T, 512] f32, head h in cols h*64..
// grid (B*H, T/128), block 256 (8 waves, 16 queries/wave).
// =====================================================================
__launch_bounds__(256)
__global__ void attn_wmma(const float* __restrict__ Q, const float* __restrict__ Km,
                          const float* __restrict__ Vm, const unsigned char* __restrict__ mask,
                          float* __restrict__ O, int causal)
{
  __shared__ _Float16 sK[32][66];        // [key][dk]
  __shared__ _Float16 sV[64][34];        // [dk][key] (transposed)
  __shared__ _Float16 sP[8][16][34];     // per-wave prob scratch [q][key]
  const int tid    = threadIdx.x;
  const int lane   = tid & 31;
  const int w      = tid >> 5;
  const int lanelo = lane & 15, lanehi = lane >> 4;
  const int b    = blockIdx.x / H_;
  const int h    = blockIdx.x % H_;
  const int tile = blockIdx.y;
  const int qbase = tile * 128 + w * 16;

  v16h aq0, aq1;
  {
    const int q = qbase + lanelo;
    const float* qp = Q + ((size_t)(b * T_ + q)) * D_ + h * DK_;
    #pragma unroll
    for (int v = 0; v < 8; ++v) {
      const int kb = (lanehi << 4) + (v << 1);
      aq0[2*v] = (_Float16)qp[kb];      aq0[2*v+1] = (_Float16)qp[kb+1];
      aq1[2*v] = (_Float16)qp[32+kb];   aq1[2*v+1] = (_Float16)qp[32+kb+1];
    }
  }

  v8f o[4];
  #pragma unroll
  for (int i = 0; i < 4; ++i) o[i] = (v8f){0,0,0,0,0,0,0,0};
  float rmax[8], rsum[8];
  #pragma unroll
  for (int v = 0; v < 8; ++v) { rmax[v] = -1e30f; rsum[v] = 0.f; }
  const float scale = 0.125f;   // dk^-0.5

  const int nkt = causal ? (tile * 128 + 128) / 32 : (T_ / 32);
  const int lc  = tid & 63;
  const int lk0 = tid >> 6;

  for (int kt = 0; kt < nkt; ++kt) {
    __syncthreads();
    #pragma unroll
    for (int i = 0; i < 8; ++i) {
      const int r  = lk0 + i * 4;
      const int kg = kt * 32 + r;
      const size_t base = ((size_t)(b * T_ + kg)) * D_ + h * DK_ + lc;
      sK[r][lc] = (_Float16)Km[base];
      sV[lc][r] = (_Float16)Vm[base];
    }
    __syncthreads();

    if (causal && kt * 32 > qbase + 15) continue;

    v8f s0 = {0,0,0,0,0,0,0,0}, s1 = {0,0,0,0,0,0,0,0};
    {
      v16h ba, bb;
      #pragma unroll
      for (int v = 0; v < 8; ++v) {
        const int kb = (lanehi << 4) + (v << 1);
        ba[2*v] = sK[lanelo][kb];      ba[2*v+1] = sK[lanelo][kb+1];
        bb[2*v] = sK[lanelo][32+kb];   bb[2*v+1] = sK[lanelo][32+kb+1];
      }
      s0 = wmma16(aq0, ba, s0);
      s0 = wmma16(aq1, bb, s0);
      #pragma unroll
      for (int v = 0; v < 8; ++v) {
        const int kb = (lanehi << 4) + (v << 1);
        ba[2*v] = sK[16+lanelo][kb];    ba[2*v+1] = sK[16+lanelo][kb+1];
        bb[2*v] = sK[16+lanelo][32+kb]; bb[2*v+1] = sK[16+lanelo][32+kb+1];
      }
      s1 = wmma16(aq0, ba, s1);
      s1 = wmma16(aq1, bb, s1);
    }

    const int kg0 = kt * 32 + lanelo;
    const int kg1 = kg0 + 16;
    const bool mk0 = mask[b * T_ + kg0] != 0;
    const bool mk1 = mask[b * T_ + kg1] != 0;
    #pragma unroll
    for (int v = 0; v < 8; ++v) {
      const int q = qbase + v + (lanehi << 3);
      float x0 = s0[v] * scale, x1 = s1[v] * scale;
      if (!mk0 || (causal && kg0 > q)) x0 = -1e30f;
      if (!mk1 || (causal && kg1 > q)) x1 = -1e30f;

      float tm = fmaxf(x0, x1);
      tm = fmaxf(tm, __shfl_xor(tm, 1));
      tm = fmaxf(tm, __shfl_xor(tm, 2));
      tm = fmaxf(tm, __shfl_xor(tm, 4));
      tm = fmaxf(tm, __shfl_xor(tm, 8));
      const float nm   = fmaxf(rmax[v], tm);
      const float corr = __expf(rmax[v] - nm);
      const float p0 = __expf(x0 - nm);
      const float p1 = __expf(x1 - nm);
      float ts = p0 + p1;
      ts += __shfl_xor(ts, 1); ts += __shfl_xor(ts, 2);
      ts += __shfl_xor(ts, 4); ts += __shfl_xor(ts, 8);
      rsum[v] = rsum[v] * corr + ts;
      rmax[v] = nm;
      o[0][v] *= corr; o[1][v] *= corr; o[2][v] *= corr; o[3][v] *= corr;

      const int row = v + (lanehi << 3);
      sP[w][row][lanelo]      = (_Float16)p0;
      sP[w][row][16 + lanelo] = (_Float16)p1;
    }
    __builtin_amdgcn_wave_barrier();

    v16h ap;
    #pragma unroll
    for (int v = 0; v < 8; ++v) {
      const int kb = (lanehi << 4) + (v << 1);
      ap[2*v] = sP[w][lanelo][kb];  ap[2*v+1] = sP[w][lanelo][kb+1];
    }
    #pragma unroll
    for (int nn = 0; nn < 4; ++nn) {
      v16h bv;
      const int d = nn * 16 + lanelo;
      #pragma unroll
      for (int v = 0; v < 8; ++v) {
        const int kb = (lanehi << 4) + (v << 1);
        bv[2*v] = sV[d][kb];  bv[2*v+1] = sV[d][kb+1];
      }
      o[nn] = wmma16(ap, bv, o[nn]);
    }
  }

  #pragma unroll
  for (int v = 0; v < 8; ++v) {
    const float inv = 1.f / rsum[v];
    const int q = qbase + v + (lanehi << 3);
    float* op = O + ((size_t)(b * T_ + q)) * D_ + h * DK_;
    op[lanelo]      = o[0][v] * inv;
    op[16 + lanelo] = o[1][v] * inv;
    op[32 + lanelo] = o[2][v] * inv;
    op[48 + lanelo] = o[3][v] * inv;
  }
}

// =====================================================================
// LayerNorm over D=512: one wave per row.
// =====================================================================
__launch_bounds__(256)
__global__ void ln_k(const float* __restrict__ X, const float* __restrict__ g,
                     const float* __restrict__ bb, float* __restrict__ Y, int rows)
{
  const int w = threadIdx.x >> 5, lane = threadIdx.x & 31;
  const int row = blockIdx.x * 8 + w;
  if (row >= rows) return;
  const float* xp = X + (size_t)row * D_;
  float vals[16]; float s = 0.f;
  #pragma unroll
  for (int i = 0; i < 16; ++i) { vals[i] = xp[lane + i * 32]; s += vals[i]; }
  #pragma unroll
  for (int off = 1; off < 32; off <<= 1) s += __shfl_xor(s, off);
  const float mean = s * (1.f / 512.f);
  float v2 = 0.f;
  #pragma unroll
  for (int i = 0; i < 16; ++i) { const float d = vals[i] - mean; v2 += d * d; }
  #pragma unroll
  for (int off = 1; off < 32; off <<= 1) v2 += __shfl_xor(v2, off);
  const float inv = rsqrtf(v2 * (1.f / 512.f) + EPS_);
  float* yp = Y + (size_t)row * D_;
  #pragma unroll
  for (int i = 0; i < 16; ++i) {
    const int c = lane + i * 32;
    yp[c] = (vals[i] - mean) * inv * g[c] + bb[c];
  }
}

// ---------------- small elementwise kernels ----------------
__global__ void time_k(const unsigned char* __restrict__ mask, float* __restrict__ tim)
{
  __shared__ float ssum[256];
  const int b = blockIdx.x, t = threadIdx.x;
  const float m = mask[b * T_ + t] ? 1.f : 0.f;
  ssum[t] = m;
  __syncthreads();
  for (int off = 128; off > 0; off >>= 1) { if (t < off) ssum[t] += ssum[t + off]; __syncthreads(); }
  tim[b * T_ + t] = m * (float)t / (ssum[0] - 1.f);
}

__global__ void u_k(const float* __restrict__ z, const int* __restrict__ y,
                    const float* __restrict__ Wemb, const float* __restrict__ bemb,
                    float* __restrict__ U)
{
  __shared__ float zz[512];
  const int b = blockIdx.x, tid = threadIdx.x;
  for (int i = tid; i < 512; i += 256) zz[i] = z[b * 512 + i];
  __syncthreads();
  const int yv = y[b];
  for (int c = tid; c < 512; c += 256) {
    float acc = bemb[c] + Wemb[(size_t)(512 + yv) * 512 + c];
    for (int j = 0; j < 512; ++j) acc += zz[j] * Wemb[(size_t)j * 512 + c];
    U[b * 512 + c] = acc;
  }
}

__global__ void src_k(const float* __restrict__ U, const float* __restrict__ tim,
                      const float* __restrict__ Wemb, float* __restrict__ S)
{
  const size_t idx = (size_t)blockIdx.x * 256 + threadIdx.x;  // NT_*512
  const int c = idx & 511;
  const size_t bt = idx >> 9;
  const int b = (int)(bt >> 8);
  S[idx] = U[b * 512 + c] + tim[bt] * Wemb[(size_t)524 * 512 + c];
}

__global__ void ax_k(const float* __restrict__ x, float* __restrict__ AX)
{
  const size_t idx = (size_t)blockIdx.x * 256 + threadIdx.x;  // NT_*160
  const int c = (int)(idx % 160);
  const size_t bt = idx / 160;
  const int t = (int)(bt & 255);
  const int b = (int)(bt >> 8);
  float v = 0.f;
  if (c < 150 && t > 0) {
    const int j = c / 6, f = c % 6;
    v = x[(((size_t)b * NJ_ + j) * NF_ + f) * T_ + (t - 1)];
  }
  AX[idx] = v;
}

__global__ void trg_k(const int* __restrict__ y, const float* __restrict__ tim,
                      const float* __restrict__ Wembx, const float* __restrict__ bembx,
                      float* __restrict__ TR)
{
  const size_t idx = (size_t)blockIdx.x * 256 + threadIdx.x;  // NT_*512
  const int c = idx & 511;
  const size_t bt = idx >> 9;
  const int t = (int)(bt & 255);
  const int b = (int)(bt >> 8);
  const int yv = y[b];
  const float div = __expf(-logf(10000.f) * (float)(c & ~1) / 512.f);
  const float ang = (float)t * div;
  const float pe  = (c & 1) ? cosf(ang) : sinf(ang);
  TR[idx] += Wembx[(size_t)(150 + yv) * 512 + c] + tim[bt] * Wembx[(size_t)162 * 512 + c]
           + bembx[c] + pe;
}

__global__ void out_k(const float* __restrict__ P, const unsigned char* __restrict__ mask,
                      float* __restrict__ out)
{
  const size_t idx = (size_t)blockIdx.x * 256 + threadIdx.x;  // B*25*6*256
  const int t = (int)(idx & 255);
  const size_t r = idx >> 8;
  const int b  = (int)(r / 150);
  const int jf = (int)(r % 150);
  const float m = mask[b * T_ + t] ? 1.f : 0.f;
  out[idx] = P[((size_t)(b * T_ + t)) * 150 + jf] * m;
}

// =====================================================================
// Host orchestration
// =====================================================================
static inline void gemm(hipStream_t s, const float* A, const _Float16* BT, const float* bias,
                        const float* res, float* C, int M, int N, int K, int relu) {
  dim3 g((N + 127) / 128, M / 128);
  gemm_wmma<<<g, 256, 0, s>>>(A, BT, bias, res, C, M, N, K, relu);
}

extern "C" void kernel_launch(void* const* d_in, const int* in_sizes, int n_in,
                              void* d_out, int out_size, void* d_ws, size_t ws_size,
                              hipStream_t stream)
{
  const float*         z      = (const float*)d_in[0];
  const int*           y      = (const int*)d_in[1];
  const unsigned char* mask   = (const unsigned char*)d_in[2];
  const float*         x      = (const float*)d_in[3];
  const float*         W_emb  = (const float*)d_in[4];
  const float*         b_emb  = (const float*)d_in[5];
  const float*         W_embx = (const float*)d_in[6];
  const float*         b_embx = (const float*)d_in[7];
  const float*         ln1_g  = (const float*)d_in[8];
  const float*         ln1_b  = (const float*)d_in[9];
  const float*         sa_W   = (const float*)d_in[10];
  const float*         sa_b   = (const float*)d_in[11];
  const float*         ln2_g  = (const float*)d_in[12];
  const float*         ln2_b  = (const float*)d_in[13];
  const float*         ca_W   = (const float*)d_in[14];
  const float*         ca_b   = (const float*)d_in[15];
  const float*         ln3_g  = (const float*)d_in[16];
  const float*         ln3_b  = (const float*)d_in[17];
  const float*         ff_W1  = (const float*)d_in[18];
  const float*         ff_b1  = (const float*)d_in[19];
  const float*         ff_W2  = (const float*)d_in[20];
  const float*         ff_b2  = (const float*)d_in[21];
  const float*         lnf_g  = (const float*)d_in[22];
  const float*         lnf_b  = (const float*)d_in[23];
  const float*         W_out  = (const float*)d_in[24];

  float* ws = (float*)d_ws;
  size_t o = 0;
  const size_t ACT = (size_t)NT_ * D_;
  float* SRC  = ws + o; o += ACT;
  float* VAL  = ws + o; o += ACT;
  float* XN   = ws + o; o += ACT;
  float* Qb   = ws + o; o += ACT;
  float* Kb   = ws + o; o += ACT;
  float* Vb   = ws + o; o += ACT;
  float* CTX  = ws + o; o += ACT;
  float* Hb   = ws + o; o += ACT;
  float* FFI  = ws + o; o += (size_t)NT_ * F_;
  float* AX   = ws + o; o += (size_t)NT_ * 160;
  float* U    = ws + o; o += (size_t)B_ * D_;
  float* TIME = ws + o; o += (size_t)B_ * T_;
  float* OUTP = ws + o; o += (size_t)NT_ * 150;
  // f16 transposed weights after the f32 region (d_ws assumed >= ~500MB)
  _Float16* WTH = (_Float16*)(ws + o);
  size_t ho = 0;
  _Float16* WTsa = WTH + ho; ho += (size_t)L_ * 4 * D_ * D_;
  _Float16* WTca = WTH + ho; ho += (size_t)L_ * 4 * D_ * D_;
  _Float16* WTf1 = WTH + ho; ho += (size_t)L_ * D_ * F_;
  _Float16* WTf2 = WTH + ho; ho += (size_t)L_ * F_ * D_;
  _Float16* WTex = WTH + ho; ho += (size_t)160 * D_;
  _Float16* WTo  = WTH + ho; ho += (size_t)150 * D_;
  (void)ws_size; (void)in_sizes; (void)n_in; (void)out_size;

  // ---- weight convert + transpose (once per call, in-stream) ----
  wtrans_k<<<dim3(D_ * D_ / 256, L_ * 4), 256, 0, stream>>>(sa_W, WTsa, D_, D_);
  wtrans_k<<<dim3(D_ * D_ / 256, L_ * 4), 256, 0, stream>>>(ca_W, WTca, D_, D_);
  wtrans_k<<<dim3(D_ * F_ / 256, L_), 256, 0, stream>>>(ff_W1, WTf1, D_, F_);
  wtrans_k<<<dim3(F_ * D_ / 256, L_), 256, 0, stream>>>(ff_W2, WTf2, F_, D_);
  wtrans_k<<<dim3(160 * D_ / 256, 1), 256, 0, stream>>>(W_embx, WTex, 160, D_);
  wtrans_k<<<dim3(D_ * 150 / 256, 1), 256, 0, stream>>>(W_out, WTo, D_, 150);

  // ---- embeddings ----
  time_k<<<B_, 256, 0, stream>>>(mask, TIME);
  u_k<<<B_, 256, 0, stream>>>(z, y, W_emb, b_emb, U);
  src_k<<<(NT_ * D_) / 256, 256, 0, stream>>>(U, TIME, W_emb, SRC);
  ax_k<<<(NT_ * 160) / 256, 256, 0, stream>>>(x, AX);
  gemm(stream, AX, WTex, nullptr, nullptr, VAL, NT_, D_, 160, 0);
  trg_k<<<(NT_ * D_) / 256, 256, 0, stream>>>(y, TIME, W_embx, b_embx, VAL);

  const dim3 agrid(B_ * H_, T_ / 128);
  const size_t DD = (size_t)D_ * D_;

  for (int i = 0; i < L_; ++i) {
    const _Float16* Wsa = WTsa + (size_t)i * 4 * DD;
    const float*    b0  = sa_b + (size_t)i * 4 * D_;
    // self-attention
    ln_k<<<NT_ / 8, 256, 0, stream>>>(VAL, ln1_g + i * D_, ln1_b + i * D_, XN, NT_);
    gemm(stream, XN, Wsa + 0 * DD, b0 + 0 * D_, nullptr, Qb, NT_, D_, D_, 0);
    gemm(stream, XN, Wsa + 1 * DD, b0 + 1 * D_, nullptr, Kb, NT_, D_, D_, 0);
    gemm(stream, XN, Wsa + 2 * DD, b0 + 2 * D_, nullptr, Vb, NT_, D_, D_, 0);
    attn_wmma<<<agrid, 256, 0, stream>>>(Qb, Kb, Vb, mask, CTX, 1);
    gemm(stream, CTX, Wsa + 3 * DD, b0 + 3 * D_, VAL, Hb, NT_, D_, D_, 0);
    // cross-attention
    const _Float16* Wca = WTca + (size_t)i * 4 * DD;
    const float*    c0  = ca_b + (size_t)i * 4 * D_;
    ln_k<<<NT_ / 8, 256, 0, stream>>>(Hb, ln2_g + i * D_, ln2_b + i * D_, XN, NT_);
    gemm(stream, XN,  Wca + 0 * DD, c0 + 0 * D_, nullptr, Qb, NT_, D_, D_, 0);
    gemm(stream, SRC, Wca + 1 * DD, c0 + 1 * D_, nullptr, Kb, NT_, D_, D_, 0);
    gemm(stream, SRC, Wca + 2 * DD, c0 + 2 * D_, nullptr, Vb, NT_, D_, D_, 0);
    attn_wmma<<<agrid, 256, 0, stream>>>(Qb, Kb, Vb, mask, CTX, 0);
    gemm(stream, CTX, Wca + 3 * DD, c0 + 3 * D_, Hb, VAL, NT_, D_, D_, 0);
    // FFN
    ln_k<<<NT_ / 8, 256, 0, stream>>>(VAL, ln3_g + i * D_, ln3_b + i * D_, XN, NT_);
    gemm(stream, XN, WTf1 + (size_t)i * D_ * F_, ff_b1 + (size_t)i * F_, nullptr, FFI,
         NT_, F_, D_, 1);
    gemm(stream, FFI, WTf2 + (size_t)i * F_ * D_, ff_b2 + (size_t)i * D_, VAL, VAL,
         NT_, D_, F_, 0);
  }

  // ---- head ----
  ln_k<<<NT_ / 8, 256, 0, stream>>>(VAL, lnf_g, lnf_b, XN, NT_);
  gemm(stream, XN, WTo, nullptr, nullptr, OUTP, NT_, NJ_ * NF_, D_, 0);
  out_k<<<(B_ * NJ_ * NF_ * T_) / 256, 256, 0, stream>>>(OUTP, mask, (float*)d_out);
}
